// MultiHeadAttention_68959994904763
// MI455X (gfx1250) — compile-verified, matrix operand order checked by reading, not run on hardware
//
#include <hip/hip_runtime.h>

typedef _Float16 h16;
typedef __attribute__((ext_vector_type(16))) _Float16 v16h;
typedef __attribute__((ext_vector_type(8)))  _Float16 v8h;
typedef __attribute__((ext_vector_type(8)))  float    v8f;

#define NB 2
#define NS 2048
#define NE 1024
#define NH 16
#define ND 64
#define ROWS (NB * NS) /* 4096 */

static __device__ __forceinline__ v8f wmma_f16(v16h a, v16h b, v8f c) {
  // 8 args: (neg_a, A, neg_b, B, c_mod, C, reuse_a, reuse_b)
  return __builtin_amdgcn_wmma_f32_16x16x32_f16(false, a, false, b, (short)0, c,
                                                false, false);
}

// Two contiguous 16-byte loads -> one 16-half fragment (b128 pairs)
static __device__ __forceinline__ v16h load_frag2(const h16* p0, const h16* p1) {
  v8h lo = *(const v8h*)p0;
  v8h hi = *(const v8h*)p1;
  return __builtin_shufflevector(lo, hi, 0, 1, 2, 3, 4, 5, 6, 7, 8, 9, 10, 11,
                                 12, 13, 14, 15);
}

__global__ void cvt_kernel(const float* __restrict__ src, h16* __restrict__ dst,
                           int n) {
  int i = blockIdx.x * blockDim.x + threadIdx.x;
  if (i < n) dst[i] = (h16)src[i];
}

// Batched transpose-convert: src [batch, R, C] f32 -> dst [batch, C, R] f16
__global__ void cvt_t_kernel(const float* __restrict__ src,
                             h16* __restrict__ dst, int R, int C, int n) {
  int i = blockIdx.x * blockDim.x + threadIdx.x;
  if (i >= n) return;
  const int rc = R * C;
  const int bb = i / rc;
  const int rem = i - bb * rc;
  const int r = rem / C;
  const int c = rem - r * C;
  dst[(size_t)bb * rc + (size_t)c * R + r] = (h16)src[i];
}

// Per-head projection: out = x @ W[h] + bias[h], W pre-transposed [H,D,E].
// grid = (ROWS/128, NH), block = 128 (4 waves, each wave -> 32 rows x 64 cols)
// vt=0: out[b,h,s,d];  vt=1: out[b,h,d,s] (transposed, for V)
__global__ __launch_bounds__(128) void qkv_gemm_kernel(
    const h16* __restrict__ X,       // [ROWS, NE]
    const h16* __restrict__ Wt,      // [NH, ND, NE]
    const float* __restrict__ bias,  // [NH, ND]
    h16* __restrict__ out, int vt)
{
  const int tid  = threadIdx.x;
  const int wave = tid >> 5;
  const int lane = tid & 31;
  const int half = lane >> 4;
  const int l16  = lane & 15;
  const int kb0  = half * 8;
  const int kb1  = 16 + half * 8;
  const int h    = blockIdx.y;
  const int rowbase = blockIdx.x * 128 + wave * 32;

  const h16* Wh = Wt + (size_t)h * ND * NE;
  v8f acc[2][4] = {};

  for (int kc = 0; kc < NE; kc += 32) {
    const h16* xr0 = X + (size_t)(rowbase + l16) * NE + kc;
    const h16* xr1 = X + (size_t)(rowbase + 16 + l16) * NE + kc;
    // hoist ALL loads: 4 B fragments + 2 A fragments live at once so the
    // scheduler can issue partial s_wait_loadcnt and overlap loads with WMMAs
    v16h a0 = load_frag2(xr0 + kb0, xr0 + kb1);
    v16h a1 = load_frag2(xr1 + kb0, xr1 + kb1);
    v16h bm[4];
#pragma unroll
    for (int nt = 0; nt < 4; ++nt) {
      const h16* wr = Wh + (size_t)(nt * 16 + l16) * NE + kc;
      bm[nt] = load_frag2(wr + kb0, wr + kb1);
    }
    if (kc + 32 < NE) {
      __builtin_prefetch(xr0 + 32, 0, 3);
      __builtin_prefetch(xr1 + 32, 0, 3);
    }
#pragma unroll
    for (int nt = 0; nt < 4; ++nt) {
      acc[0][nt] = wmma_f16(a0, bm[nt], acc[0][nt]);
      acc[1][nt] = wmma_f16(a1, bm[nt], acc[1][nt]);
    }
  }

#pragma unroll
  for (int mt = 0; mt < 2; ++mt)
#pragma unroll
    for (int nt = 0; nt < 4; ++nt)
#pragma unroll
      for (int r = 0; r < 8; ++r) {
        const int rowg = rowbase + mt * 16 + half * 8 + r;
        const int b    = rowg >> 11;  // / NS
        const int s    = rowg & (NS - 1);
        const int n    = nt * 16 + l16;
        const float v  = acc[mt][nt][r] + bias[h * ND + n];
        if (vt)
          out[((size_t)(b * NH + h) * ND + n) * NS + s] = (h16)v;
        else
          out[((size_t)(b * NH + h) * NS + s) * ND + n] = (h16)v;
      }
}

// Flash-attention: one wave per 32-row query tile per (b,h); K/V fragments
// are shared by both 16-row sub-tiles (2x WMMA per byte loaded from L2).
// grid = NB*NH*(NS/32), block = 32.  V is transposed [NB*NH, ND, NS].
__global__ __launch_bounds__(32) void attn_kernel(
    const h16* __restrict__ Q,   // [NB*NH, NS, ND]
    const h16* __restrict__ K,   // [NB*NH, NS, ND]
    const h16* __restrict__ Vt,  // [NB*NH, ND, NS]
    h16* __restrict__ O)         // [NB, NS, NH*ND]
{
  __shared__ h16 lp[2][16 * 32];  // P tile relayout buffers (one per m-tile)
  const int lane = threadIdx.x;
  const int half = lane >> 4;
  const int l16  = lane & 15;
  const int kb0  = half * 8;
  const int kb1  = 16 + half * 8;
  const int qt   = blockIdx.x & (NS / 32 - 1);
  const int bh   = blockIdx.x >> 6;
  const int h    = bh & (NH - 1);
  const int b    = bh >> 4;
  const int qbase = qt * 32;

  const h16* Qp = Q + (size_t)bh * NS * ND;
  const h16* Kp = K + (size_t)bh * NS * ND;
  const h16* Vp = Vt + (size_t)bh * ND * NS;

  // Q fragments: 2 m-tiles x 2 K-chunks, resident for the whole pass
  v16h aq[2][2];
#pragma unroll
  for (int mt = 0; mt < 2; ++mt) {
    const h16* qrow = Qp + (size_t)(qbase + mt * 16 + l16) * ND;
#pragma unroll
    for (int kc = 0; kc < 2; ++kc)
      aq[mt][kc] = load_frag2(qrow + kc * 32 + kb0, qrow + kc * 32 + kb1);
  }

  v8f acc[2][4] = {};
  float row_m[2][8], row_l[2][8];
#pragma unroll
  for (int mt = 0; mt < 2; ++mt)
#pragma unroll
    for (int r = 0; r < 8; ++r) { row_m[mt][r] = -1e30f; row_l[mt][r] = 0.0f; }

  const float scale  = 0.125f;        // 1/sqrt(64)
  const int nchunks = (qbase >> 5) + 1;  // causal bound (rows < qbase+32)

  for (int ch = 0; ch < nchunks; ++ch) {
    const int tbase = ch * 32;

    // ---- K fragments (hoisted; shared by both m-tiles) ----
    v16h bk[2][2];  // [tt][kc]
#pragma unroll
    for (int tt = 0; tt < 2; ++tt) {
      const h16* krow = Kp + (size_t)(tbase + tt * 16 + l16) * ND;
#pragma unroll
      for (int kc = 0; kc < 2; ++kc)
        bk[tt][kc] = load_frag2(krow + kc * 32 + kb0, krow + kc * 32 + kb1);
    }
    if (ch + 1 < nchunks) {
      __builtin_prefetch(Kp + (size_t)(tbase + 32 + l16) * ND, 0, 3);
      __builtin_prefetch(Kp + (size_t)(tbase + 48 + l16) * ND, 0, 3);
    }

    // ---- scores: S = Q K^T ----
    v8f c[2][2] = {};  // [mt][tt]
#pragma unroll
    for (int mt = 0; mt < 2; ++mt)
#pragma unroll
      for (int tt = 0; tt < 2; ++tt)
#pragma unroll
        for (int kc = 0; kc < 2; ++kc)
          c[mt][tt] = wmma_f16(aq[mt][kc], bk[tt][kc], c[mt][tt]);

    // ---- scale + causal mask + online softmax (per m-tile) ----
    float corr[2][8];
#pragma unroll
    for (int mt = 0; mt < 2; ++mt) {
#pragma unroll
      for (int r = 0; r < 8; ++r) {
        const int srow = qbase + mt * 16 + half * 8 + r;
        float s0 = c[mt][0][r] * scale;
        float s1 = c[mt][1][r] * scale;
        if (tbase + l16 > srow)      s0 = -1e30f;
        if (tbase + 16 + l16 > srow) s1 = -1e30f;
        float mx = fmaxf(s0, s1);
#pragma unroll
        for (int o = 1; o < 16; o <<= 1) mx = fmaxf(mx, __shfl_xor(mx, o, 32));
        const float mnew = fmaxf(row_m[mt][r], mx);
        const float p0 = __expf(s0 - mnew);
        const float p1 = __expf(s1 - mnew);
        corr[mt][r]  = __expf(row_m[mt][r] - mnew);
        row_m[mt][r] = mnew;
        float ps = p0 + p1;
#pragma unroll
        for (int o = 1; o < 16; o <<= 1) ps += __shfl_xor(ps, o, 32);
        row_l[mt][r] = row_l[mt][r] * corr[mt][r] + ps;
        lp[mt][(half * 8 + r) * 32 + l16]      = (h16)p0;
        lp[mt][(half * 8 + r) * 32 + 16 + l16] = (h16)p1;
      }
    }
    __syncthreads();

    // rescale running accumulators
#pragma unroll
    for (int mt = 0; mt < 2; ++mt)
#pragma unroll
      for (int nt = 0; nt < 4; ++nt)
#pragma unroll
        for (int r = 0; r < 8; ++r) acc[mt][nt][r] *= corr[mt][r];

    // P fragments (C-layout -> A-layout via LDS), contiguous b128 reads
    v16h ap[2];
#pragma unroll
    for (int mt = 0; mt < 2; ++mt)
      ap[mt] = load_frag2(&lp[mt][l16 * 32 + kb0], &lp[mt][l16 * 32 + kb1]);

    // ---- V fragments (hoisted; shared by both m-tiles) ----
    v16h bv[4];
#pragma unroll
    for (int nt = 0; nt < 4; ++nt) {
      const h16* vr = Vp + (size_t)(nt * 16 + l16) * NS + tbase;
      bv[nt] = load_frag2(vr + kb0, vr + kb1);
    }

    // ---- O += P V ----
#pragma unroll
    for (int nt = 0; nt < 4; ++nt) {
      acc[0][nt] = wmma_f16(ap[0], bv[nt], acc[0][nt]);
      acc[1][nt] = wmma_f16(ap[1], bv[nt], acc[1][nt]);
    }
    __syncthreads();
  }

  // normalize + store (head-concat layout [B, S, H*D])
#pragma unroll
  for (int mt = 0; mt < 2; ++mt)
#pragma unroll
    for (int r = 0; r < 8; ++r) {
      const float inv = 1.0f / row_l[mt][r];
      const int s = qbase + mt * 16 + half * 8 + r;
#pragma unroll
      for (int nt = 0; nt < 4; ++nt) {
        O[((size_t)b * NS + s) * (NH * ND) + h * ND + nt * 16 + l16] =
            (h16)(acc[mt][nt][r] * inv);
      }
    }
}

// Output projection: out = A @ W_o + b_o (f32), W_o pre-transposed [NE, NH*ND]
// grid = (ROWS/128, NE/64), block = 128 (4 waves, each wave 32 rows x 64 cols)
__global__ __launch_bounds__(128) void out_gemm_kernel(
    const h16* __restrict__ A,       // [ROWS, NH*ND]
    const h16* __restrict__ Wot,     // [NE, NH*ND]
    const float* __restrict__ bias,  // [NE]
    float* __restrict__ out)         // [ROWS, NE]
{
  const int tid  = threadIdx.x;
  const int wave = tid >> 5;
  const int lane = tid & 31;
  const int half = lane >> 4;
  const int l16  = lane & 15;
  const int kb0  = half * 8;
  const int kb1  = 16 + half * 8;
  const int rowbase = blockIdx.x * 128 + wave * 32;
  const int ntbase  = blockIdx.y * 64;
  const int KDIM = NH * ND;  // 1024

  v8f acc[2][4] = {};
  for (int kc = 0; kc < KDIM; kc += 32) {
    const h16* ar0 = A + (size_t)(rowbase + l16) * KDIM + kc;
    const h16* ar1 = A + (size_t)(rowbase + 16 + l16) * KDIM + kc;
    v16h a0 = load_frag2(ar0 + kb0, ar0 + kb1);
    v16h a1 = load_frag2(ar1 + kb0, ar1 + kb1);
    v16h bm[4];
#pragma unroll
    for (int nt = 0; nt < 4; ++nt) {
      const h16* wr = Wot + (size_t)(ntbase + nt * 16 + l16) * KDIM + kc;
      bm[nt] = load_frag2(wr + kb0, wr + kb1);
    }
    if (kc + 32 < KDIM) {
      __builtin_prefetch(ar0 + 32, 0, 3);
      __builtin_prefetch(ar1 + 32, 0, 3);
    }
#pragma unroll
    for (int nt = 0; nt < 4; ++nt) {
      acc[0][nt] = wmma_f16(a0, bm[nt], acc[0][nt]);
      acc[1][nt] = wmma_f16(a1, bm[nt], acc[1][nt]);
    }
  }

#pragma unroll
  for (int mt = 0; mt < 2; ++mt)
#pragma unroll
    for (int nt = 0; nt < 4; ++nt)
#pragma unroll
      for (int r = 0; r < 8; ++r) {
        const int rowg = rowbase + mt * 16 + half * 8 + r;
        const int n    = ntbase + nt * 16 + l16;
        out[(size_t)rowg * NE + n] = acc[mt][nt][r] + bias[n];
      }
}

extern "C" void kernel_launch(void* const* d_in, const int* in_sizes, int n_in,
                              void* d_out, int out_size, void* d_ws,
                              size_t ws_size, hipStream_t stream) {
  const float* x   = (const float*)d_in[0];
  const float* W_q = (const float*)d_in[1];
  const float* b_q = (const float*)d_in[2];
  const float* W_k = (const float*)d_in[3];
  const float* b_k = (const float*)d_in[4];
  const float* W_v = (const float*)d_in[5];
  const float* b_v = (const float*)d_in[6];
  const float* W_o = (const float*)d_in[7];
  const float* b_o = (const float*)d_in[8];
  float* out = (float*)d_out;

  const size_t nx   = (size_t)ROWS * NE;          // 4M halves
  const size_t nqkv = (size_t)NB * NH * NS * ND;  // 4M halves
  const size_t nw   = (size_t)NH * NE * ND;       // 1M halves
  h16* xh  = (h16*)d_ws;
  h16* qh  = xh + nx;
  h16* kh  = qh + nqkv;
  h16* vh  = kh + nqkv;   // transposed V: [NB*NH, ND, NS]
  h16* oh  = vh + nqkv;
  h16* wqh = oh + nx;     // transposed: [NH, ND, NE]
  h16* wkh = wqh + nw;
  h16* wvh = wkh + nw;
  h16* woh = wvh + nw;    // transposed: [NE, NH*ND]

  cvt_kernel<<<(int)((nx + 255) / 256), 256, 0, stream>>>(x, xh, (int)nx);
  cvt_t_kernel<<<(int)((nw + 255) / 256), 256, 0, stream>>>(W_q, wqh, NE, ND,
                                                            (int)nw);
  cvt_t_kernel<<<(int)((nw + 255) / 256), 256, 0, stream>>>(W_k, wkh, NE, ND,
                                                            (int)nw);
  cvt_t_kernel<<<(int)((nw + 255) / 256), 256, 0, stream>>>(W_v, wvh, NE, ND,
                                                            (int)nw);
  cvt_t_kernel<<<(int)((nw + 255) / 256), 256, 0, stream>>>(W_o, woh, NH * ND,
                                                            NE, (int)nw);

  dim3 gq(ROWS / 128, NH);
  qkv_gemm_kernel<<<gq, 128, 0, stream>>>(xh, wqh, b_q, qh, 0);
  qkv_gemm_kernel<<<gq, 128, 0, stream>>>(xh, wkh, b_k, kh, 0);
  qkv_gemm_kernel<<<gq, 128, 0, stream>>>(xh, wvh, b_v, vh, 1);

  attn_kernel<<<NB * NH * (NS / 32), 32, 0, stream>>>(qh, kh, vh, oh);

  dim3 go(ROWS / 128, NE / 64);
  out_gemm_kernel<<<go, 128, 0, stream>>>(oh, woh, b_o, out);
}
